// QuantizerCSS_47270410059800
// MI455X (gfx1250) — matrix operand order repaired and checked by reading, not compiled, not run on hardware
//
#include <hip/hip_runtime.h>
#include <hip/hip_bf16.h>

typedef __attribute__((ext_vector_type(16))) __bf16 bf16x16;
typedef __attribute__((ext_vector_type(8)))  float  f32x8;
typedef __attribute__((ext_vector_type(4)))  unsigned uint32x4;
typedef __attribute__((ext_vector_type(8)))  int      int32x8;
typedef __attribute__((ext_vector_type(4)))  int      int32x4;

#define N_IMG 32
#define E_DIM 512
#define T_DIM 512
#define M_TOT (N_IMG * T_DIM)   // 16384
#define NE    8192
#define ROW_TILE 64
#define COL_TILE 32
#define PITCH 520                // ushort pitch: 1024B row + 16B TDM pad (bank skew 4)
#define KSTEPS (E_DIM / 32)      // 16
#define NCHUNK (NE / COL_TILE)   // 256
#define EC_SZ (COL_TILE * PITCH) // one emb LDS buffer, in ushorts

union Frag { unsigned u[8]; bf16x16 v; };

__device__ __forceinline__ unsigned short f2bf(float f) {
  unsigned u = __float_as_uint(f);
  u += 0x7FFFu + ((u >> 16) & 1u);          // round-to-nearest-even
  return (unsigned short)(u >> 16);
}

// Issue one TDM descriptor: 2D tile of 32 codes x 1024B (bf16 row) from global
// into LDS, hardware-padded +16B per row (=> LDS pitch 1040B = 520 ushorts).
__device__ __forceinline__ void tdm_load_chunk(const unsigned short* embb,
                                               unsigned short* ldsdst, int cb) {
  unsigned long long ga =
      (unsigned long long)(uintptr_t)embb + (unsigned long long)cb * (E_DIM * 2);
  uint32x4 g0;
  g0[0] = 1u;                                   // count=1, user mode, no gather
  g0[1] = (unsigned)(uintptr_t)ldsdst;          // LDS byte address
  g0[2] = (unsigned)ga;                         // global_addr[31:0]
  g0[3] = (unsigned)(ga >> 32) | (2u << 30);    // global_addr[56:32] | type=2
  int32x8 g1;
  g1[0] = (3 << 16)        // data_size = 8B units
        | (1 << 20)        // pad_enable
        | (7 << 22)        // pad_interval = 256 dwords (1024B)
        | (3 << 25);       // pad_amount  = 4 dwords (16B)
  g1[1] = (128 << 16);     // tensor_dim0 = 128 (8B units per row)
  g1[2] = (32 << 16);      // tensor_dim1 = 32 rows
  g1[3] = (128 << 16);     // tile_dim0 = 128
  g1[4] = 32;              // tile_dim1 = 32, tile_dim2 = 0
  g1[5] = 128;             // tensor_dim0_stride = 128 (contiguous rows)
  g1[6] = 0;
  g1[7] = 0;
  int32x4 zz4 = {0, 0, 0, 0};
  int32x8 zz8 = {0, 0, 0, 0, 0, 0, 0, 0};
  __builtin_amdgcn_tensor_load_to_lds(g0, g1, zz4, zz4, zz8, 0);
}

// ---------------- emb -> bf16 + ||e||^2 ----------------
__global__ void prep_emb(const float* __restrict__ emb,
                         unsigned short* __restrict__ embb,
                         float* __restrict__ enorm) {
  int j = blockIdx.x;
  int tid = threadIdx.x;
  float s = 0.f;
  for (int e = tid; e < E_DIM; e += 256) {
    float v = emb[(size_t)j * E_DIM + e];
    embb[(size_t)j * E_DIM + e] = f2bf(v);
    s += v * v;
  }
  for (int mask = 16; mask >= 1; mask >>= 1) s += __shfl_xor(s, mask, 32);
  __shared__ float ws[8];
  if ((tid & 31) == 0) ws[tid >> 5] = s;
  __syncthreads();
  if (tid == 0) {
    float t = 0.f;
    for (int i = 0; i < 8; ++i) t += ws[i];
    enorm[j] = t;
  }
}

__global__ void init_stats(float* __restrict__ counts, float* __restrict__ acc) {
  int i = blockIdx.x * blockDim.x + threadIdx.x;
  if (i < NE) counts[i] = 0.f;
  if (i == 0) acc[0] = 0.f;
}

// ------- fused bf16-WMMA GEMM + argmin, TDM quad-buffer, 1 barrier/chunk -------
__global__ void __launch_bounds__(256, 1)
nn_search(const float* __restrict__ z, const unsigned short* __restrict__ embb,
          const float* __restrict__ enorm, int* __restrict__ bidx,
          float* __restrict__ counts) {
  extern __shared__ unsigned short smem[];
  unsigned short* zf = smem;                                // ROW_TILE * PITCH
  unsigned short* ec = smem + ROW_TILE * PITCH;             // 4 * EC_SZ
  float* mg = (float*)(ec + 4 * EC_SZ);                     // 64 x 2 x {v,i}

  int tid  = threadIdx.x;
  int lane = tid & 31, wv = tid >> 5;
  int rg = wv >> 1, cg = wv & 1;                            // 4 row-groups x 2 col-groups
  int m0 = blockIdx.x * ROW_TILE;
  int n  = m0 / T_DIM;
  int t0 = m0 % T_DIM;

  // kick off TDM for chunks 0,1 while we stage/convert the z tile
  if (wv == 0) {
    tdm_load_chunk(embb, ec, 0);
    tdm_load_chunk(embb, ec + EC_SZ, 1);
  }

  // stage zf tile: transpose [E,T] slab -> LDS [t][e], convert to bf16
  for (int i = tid; i < ROW_TILE * E_DIM; i += 256) {
    int tl = i & (ROW_TILE - 1);
    int e  = i >> 6;
    float v = z[((size_t)n * E_DIM + e) * T_DIM + t0 + tl];
    zf[tl * PITCH + e] = f2bf(v);
  }
  __syncthreads();

  // build all A fragments once (reused for all 8192 codes) — kept in VGPRs
  int mrow = rg * 16 + (lane & 15);
  int hb   = lane >> 4;
  const unsigned short* arow = zf + mrow * PITCH;
  Frag afrag[KSTEPS];
#pragma unroll
  for (int ks = 0; ks < KSTEPS; ++ks) {
    int kb = ks * 32;
#pragma unroll
    for (int j = 0; j < 8; ++j) {
      int koff = 16 * hb + 2 * j;
      afrag[ks].u[j] = *(const unsigned*)(arow + kb + koff);
    }
  }

  float bv[8];
  int   bi[8];
#pragma unroll
  for (int r = 0; r < 8; ++r) { bv[r] = 3.4e38f; bi[r] = 0; }

  int boff = (cg * 16 + (lane & 15)) * PITCH + hb * 16;
  int ncol_local = cg * 16 + (lane & 15);

  for (int it = 0; it < NCHUNK; ++it) {
    // chunk it+2 overwrites buf[(it-2)&3]; all waves finished compute(it-2)
    // before signaling barrier #(it-1), which we already passed.
    if (wv == 0) {
      if (it + 2 < NCHUNK) {
        tdm_load_chunk(embb, ec + ((it + 2) & 3) * EC_SZ, it + 2);
        __builtin_amdgcn_s_wait_tensorcnt(2);   // chunk `it` landed
      } else if (it + 1 < NCHUNK) {
        __builtin_amdgcn_s_wait_tensorcnt(1);
      } else {
        __builtin_amdgcn_s_wait_tensorcnt(0);
      }
    }
    __syncthreads();   // publish chunk `it`; compute(it-1) globally done

    const unsigned short* brow = ec + (it & 3) * EC_SZ + boff;
    f32x8 c0 = {0.f, 0.f, 0.f, 0.f, 0.f, 0.f, 0.f, 0.f};
    f32x8 c1 = {0.f, 0.f, 0.f, 0.f, 0.f, 0.f, 0.f, 0.f};

    // software-pipelined B fragments: load ks+1 while WMMA consumes ks
    Frag bb[2];
#pragma unroll
    for (int j = 0; j < 8; ++j)
      bb[0].u[j] = *(const unsigned*)(brow + 2 * j);
#pragma unroll
    for (int ks = 0; ks < KSTEPS; ++ks) {
      if (ks + 1 < KSTEPS) {
#pragma unroll
        for (int j = 0; j < 8; ++j)
          bb[(ks + 1) & 1].u[j] = *(const unsigned*)(brow + (ks + 1) * 32 + 2 * j);
      }
      if (ks & 1)
        c1 = __builtin_amdgcn_wmma_f32_16x16x32_bf16(false, afrag[ks].v, false,
                                                     bb[1].v, (short)0, c1,
                                                     false, false);
      else
        c0 = __builtin_amdgcn_wmma_f32_16x16x32_bf16(false, afrag[ks].v, false,
                                                     bb[0].v, (short)0, c0,
                                                     false, false);
    }
    // scheduler directive: keep 2 fragments (8 ds_read) in flight ahead of each WMMA
    __builtin_amdgcn_sched_group_barrier(0x100, 8, 0);       // 8x DS read
#pragma unroll
    for (int k = 0; k < 14; ++k) {
      __builtin_amdgcn_sched_group_barrier(0x008, 1, 0);     // 1x WMMA
      __builtin_amdgcn_sched_group_barrier(0x100, 4, 0);     // 4x DS read
    }
    __builtin_amdgcn_sched_group_barrier(0x008, 2, 0);       // final 2x WMMA

    int ncol = it * COL_TILE + ncol_local;
    float en = enorm[ncol];
#pragma unroll
    for (int r = 0; r < 8; ++r) {
      float v = en - 2.0f * (c0[r] + c1[r]);   // ||e||^2 - 2 z.e
      if (v < bv[r]) { bv[r] = v; bi[r] = ncol; }   // strict < => first-min index
    }
  }

  // argmin reduce across the 16 lanes sharing each row half
#pragma unroll
  for (int r = 0; r < 8; ++r) {
#pragma unroll
    for (int mask = 8; mask >= 1; mask >>= 1) {
      float ov = __shfl_xor(bv[r], mask, 32);
      int   oi = __shfl_xor(bi[r], mask, 32);
      if (ov < bv[r] || (ov == bv[r] && oi < bi[r])) { bv[r] = ov; bi[r] = oi; }
    }
  }
  __syncthreads();
  if (lane == 0 || lane == 16) {
#pragma unroll
    for (int r = 0; r < 8; ++r) {
      int row = rg * 16 + hb * 8 + r;       // C layout: vgpr r -> M = r + 8*half
      mg[(row * 2 + cg) * 2 + 0] = bv[r];
      ((int*)mg)[(row * 2 + cg) * 2 + 1] = bi[r];
    }
  }
  __syncthreads();
  if (tid < ROW_TILE) {
    float v0 = mg[(tid * 2 + 0) * 2 + 0];
    float v1 = mg[(tid * 2 + 1) * 2 + 0];
    int   i0 = ((int*)mg)[(tid * 2 + 0) * 2 + 1];
    int   i1 = ((int*)mg)[(tid * 2 + 1) * 2 + 1];
    int best = (v1 < v0 || (v1 == v0 && i1 < i0)) ? i1 : i0;
    bidx[m0 + tid] = best;
    atomicAdd(&counts[best], 1.0f);
  }
}

// ---------------- gather z_q (exact fp32 emb rows) + fused loss ----------------
__global__ void gather_loss(const float* __restrict__ z, const float* __restrict__ emb,
                            const int* __restrict__ bidx, float* __restrict__ out,
                            float* __restrict__ acc) {
  __shared__ int   idxs[64];
  __shared__ float tile[32][65];
  __shared__ float wsum[8];
  int tid = threadIdx.x;
  int b = blockIdx.x;
  int n = b >> 3;
  int t0 = (b & 7) * 64;
  if (tid < 64) idxs[tid] = bidx[n * T_DIM + t0 + tid];
  __syncthreads();
  float lacc = 0.f;
  for (int e0 = 0; e0 < E_DIM; e0 += 32) {
#pragma unroll
    for (int p = 0; p < 8; ++p) {
      int tl = (tid >> 5) + p * 8;
      tile[tid & 31][tl] = emb[(size_t)idxs[tl] * E_DIM + e0 + (tid & 31)];
    }
    __syncthreads();
#pragma unroll
    for (int p = 0; p < 8; ++p) {
      int el = (tid >> 6) * 8 + p;
      float q = tile[el][tid & 63];
      size_t o = ((size_t)n * E_DIM + e0 + el) * T_DIM + t0 + (tid & 63);
      out[o] = q;
      float d = q - z[o];
      lacc += d * d;
    }
    __syncthreads();
  }
  for (int mask = 16; mask >= 1; mask >>= 1) lacc += __shfl_xor(lacc, mask, 32);
  if ((tid & 31) == 0) wsum[tid >> 5] = lacc;
  __syncthreads();
  if (tid == 0) {
    float s = 0.f;
    for (int i = 0; i < 8; ++i) s += wsum[i];
    atomicAdd(acc, s);
  }
}

__global__ void finalize(const float* __restrict__ counts, const float* __restrict__ acc,
                         float* __restrict__ out) {
  __shared__ float red[256];
  int tid = threadIdx.x;
  float s = 0.f;
  for (int j = tid; j < NE; j += 256) {
    float em = counts[j] * (1.0f / (float)M_TOT);
    s += em * logf(em + 1e-10f);
  }
  red[tid] = s;
  __syncthreads();
  for (int step = 128; step >= 1; step >>= 1) {
    if (tid < step) red[tid] += red[tid + step];
    __syncthreads();
  }
  if (tid == 0) {
    size_t zoff = (size_t)N_IMG * E_DIM * T_DIM;
    out[zoff + 0] = 1.25f * acc[0] / ((float)M_TOT * (float)E_DIM);  // loss
    out[zoff + 1] = expf(-red[0]);                                   // perplexity
  }
}

extern "C" void kernel_launch(void* const* d_in, const int* in_sizes, int n_in,
                              void* d_out, int out_size, void* d_ws, size_t ws_size,
                              hipStream_t stream) {
  (void)in_sizes; (void)n_in; (void)out_size; (void)ws_size;
  const float* z   = (const float*)d_in[0];
  const float* emb = (const float*)d_in[1];
  // d_in[2] (map_w), d_in[3] (map_b): dead branch in reference — unused.
  float* out = (float*)d_out;

  char* ws = (char*)d_ws;
  unsigned short* embb = (unsigned short*)ws;                       // 8 MiB
  size_t off = (size_t)NE * E_DIM * 2;
  float* enorm  = (float*)(ws + off);         off += (size_t)NE * 4;
  int*   bidx   = (int*)  (ws + off);         off += (size_t)M_TOT * 4;
  float* counts = (float*)(ws + off);         off += (size_t)NE * 4;
  float* acc    = (float*)(ws + off);

  hipLaunchKernelGGL(prep_emb, dim3(NE), dim3(256), 0, stream, emb, embb, enorm);
  hipLaunchKernelGGL(init_stats, dim3((NE + 255) / 256), dim3(256), 0, stream, counts, acc);

  size_t lds = (size_t)(ROW_TILE * PITCH + 4 * EC_SZ) * 2 + (size_t)ROW_TILE * 2 * 8;
  hipLaunchKernelGGL(nn_search, dim3(M_TOT / ROW_TILE), dim3(256), lds, stream,
                     z, embb, enorm, bidx, counts);

  hipLaunchKernelGGL(gather_loss, dim3(N_IMG * (T_DIM / 64)), dim3(256), 0, stream,
                     z, emb, bidx, out, acc);
  hipLaunchKernelGGL(finalize, dim3(1), dim3(256), 0, stream, counts, acc, out);
}
